// GSAAttention_18734647345176
// MI455X (gfx1250) — matrix operand order repaired and checked by reading, not compiled, not run on hardware
//
#include <hip/hip_runtime.h>

typedef __attribute__((ext_vector_type(8)))  float    v8f;
typedef __attribute__((ext_vector_type(8)))  _Float16 v8h;
typedef __attribute__((ext_vector_type(16))) _Float16 v16h;

// f16 A/B-fragment helper: lane half h holds K = 16h..16h+15 (contiguous),
// packed 2/VGPR -> two contiguous v8h loads.
__device__ __forceinline__ v16h load_frag16(const _Float16* p) {
  v8h lo = *(const v8h*)(p);
  v8h hi = *(const v8h*)(p + 8);
  return __builtin_shufflevector(lo, hi, 0, 1, 2, 3, 4, 5, 6, 7,
                                 8, 9, 10, 11, 12, 13, 14, 15);
}

// ---------------------------------------------------------------------------
// Elementwise f32 -> f16 (weights).
// ---------------------------------------------------------------------------
__global__ void cvt_f16_kernel(const float* __restrict__ src,
                               _Float16* __restrict__ dst, int nElem) {
  const int i = blockIdx.x * blockDim.x + threadIdx.x;
  if (i < nElem) dst[i] = (_Float16)src[i];
}

// ---------------------------------------------------------------------------
// LDS-tiled transpose + convert: x [N][C][S] f32  ->  xT [N][S][C] f16.
// Makes every projection B-fragment a contiguous v16h load.
// ---------------------------------------------------------------------------
__global__ __launch_bounds__(256) void transpose_cvt_kernel(
    const float* __restrict__ X, _Float16* __restrict__ Xt, int C, int S)
{
  __shared__ float tile[32][33];
  const int n  = blockIdx.z;
  const int s0 = blockIdx.x * 32;
  const int c0 = blockIdx.y * 32;
  const float* Xn  = X  + (size_t)n * C * S;
  _Float16*    Xtn = Xt + (size_t)n * S * C;
  const int tx = threadIdx.x, ty = threadIdx.y;
#pragma unroll
  for (int i = 0; i < 4; ++i)
    tile[ty + 8 * i][tx] = Xn[(size_t)(c0 + ty + 8 * i) * S + (s0 + tx)];
  __syncthreads();
#pragma unroll
  for (int i = 0; i < 4; ++i)
    Xtn[(size_t)(s0 + ty + 8 * i) * C + (c0 + tx)] = (_Float16)tile[tx][ty + 8 * i];
}

// ---------------------------------------------------------------------------
// Projection GEMM on V_WMMA_F32_16X16X32_F16 (f32 accumulate + f32 bias):
//   Out[n] (Rout x S) = W16 (Rout x C) @ x[n] (C x S) + bias  -> f16
// A from W16 [Rout][C]; B from xT16 [S][C] (K-contiguous per lane).
// transposeOut=1 stores (S x Rout), r contiguous (K and V); else [Rout][S] (Q).
// ---------------------------------------------------------------------------
__global__ __launch_bounds__(256) void proj_gemm16_kernel(
    const _Float16* __restrict__ W16, const float* __restrict__ bias,
    const _Float16* __restrict__ Xt, _Float16* __restrict__ Out,
    int Rout, int C, int S, int transposeOut)
{
  const int lane = threadIdx.x & 31;
  const int wave = threadIdx.x >> 5;
  const int half = lane >> 4;
  const int l16  = lane & 15;

  const int n    = blockIdx.z;
  const int row0 = blockIdx.y * 16;
  const int col0 = blockIdx.x * 128 + wave * 16;

  const _Float16* Xn = Xt + (size_t)n * S * C;
  _Float16* On = Out + (size_t)n * Rout * S;

  const int arow = row0 + l16;
  const int bcol = col0 + l16;
  const _Float16* wrow = W16 + (size_t)arow * C + 16 * half;
  const _Float16* xcol = Xn + (size_t)bcol * C + 16 * half;

  v8f c = {};
  for (int k = 0; k < C; k += 32) {
    v16h a = load_frag16(wrow + k);
    v16h b = *(const v16h*)(xcol + k);
    c = __builtin_amdgcn_wmma_f32_16x16x32_f16(false, a, false, b, (short)0, c,
                                               false, false);
  }

#pragma unroll
  for (int v = 0; v < 8; ++v) {
    const int row = row0 + v + 8 * half;          // C/D map: M = v + 8*(lane/16)
    const _Float16 val = (_Float16)(c[v] + bias[row]);
    if (transposeOut) On[(size_t)bcol * Rout + row] = val;   // K,V: [S][R]
    else              On[(size_t)row  * S    + bcol] = val;  // Q:   [C][S]
  }
}

// ---------------------------------------------------------------------------
// Flash-style GSA attention on V_WMMA_F32_16X16X32_F16 (f32 accumulate).
//   kv[s,t] = sum_r K[n,s,r]*V[n,r,t];  att = softmax_s;  out = g*Q@att + x
// Block: one (n, 16-wide t tile), 8 waves x 64 o-rows = C. Each iteration
// consumes 32 keys: wave 0 builds two 16x16 logit tiles, runs the online
// softmax, publishes a 32x16 f16 P tile (t-major LDS) + rescale factors;
// every wave then does ONE K=32 WMMA per o-tile.
// ---------------------------------------------------------------------------
__global__ __launch_bounds__(256) void gsa_attn_kernel(
    const _Float16* __restrict__ K,   // [N][S][R]  (s-major, r contiguous)
    const _Float16* __restrict__ V,   // [N][S][R]  (t-major, r contiguous)
    const _Float16* __restrict__ Q,   // [N][C][S]
    const float* __restrict__ X,      // [N][C][S]
    const float* __restrict__ gamma,
    float* __restrict__ Out,          // [N][C][S]
    int C, int R, int S)
{
  const int lane = threadIdx.x & 31;
  const int wave = threadIdx.x >> 5;
  const int half = lane >> 4;
  const int l16  = lane & 15;
  const int wave_u = __builtin_amdgcn_readfirstlane(wave);  // scalar branch

  const int n  = blockIdx.y;
  const int t0 = blockIdx.x * 16;

  const _Float16* Kn = K + (size_t)n * S * R;
  const _Float16* Vn = V + (size_t)n * S * R;
  const _Float16* Qn = Q + (size_t)n * C * S;
  const float*    Xn = X + (size_t)n * C * S;
  float*          On = Out + (size_t)n * C * S;

  // P tile, t-major: Pl[t*40 + s], s in [0,32). Stride 40 halves keeps the
  // per-lane v16h reads 16B-aligned and bank-conflict free.
  __shared__ _Float16 Pl[16 * 40];
  __shared__ float scaleL[16];
  __shared__ float linvL[16];

  v8f acc[4] = {};                 // 64 o-rows x 16 t-cols, f32
  float m    = -INFINITY;          // running max, column t = l16 (wave 0)
  float lsum = 0.0f;

  const _Float16* vcol = Vn + (size_t)(t0 + l16) * R + 16 * half;

  const int nsb = S / 32;
  for (int sb = 0; sb < nsb; ++sb) {
    const int s0 = sb * 32;

    if (wave_u == 0) {
      v8f lc0 = {}, lc1 = {};      // logits for s rows [s0,s0+16), [s0+16,s0+32)
#pragma unroll
      for (int rb = 0; rb < 64; rb += 32) {
        v16h b  = *(const v16h*)(vcol + rb);
        v16h a0 = load_frag16(Kn + (size_t)(s0 + l16)      * R + rb + 16 * half);
        v16h a1 = load_frag16(Kn + (size_t)(s0 + 16 + l16) * R + rb + 16 * half);
        lc0 = __builtin_amdgcn_wmma_f32_16x16x32_f16(false, a0, false, b,
                                                     (short)0, lc0, false, false);
        lc1 = __builtin_amdgcn_wmma_f32_16x16x32_f16(false, a1, false, b,
                                                     (short)0, lc1, false, false);
      }
      // Online softmax over the 32 s-rows of column t = l16.
      float cmax = lc0[0];
#pragma unroll
      for (int v = 1; v < 8; ++v) cmax = fmaxf(cmax, lc0[v]);
#pragma unroll
      for (int v = 0; v < 8; ++v) cmax = fmaxf(cmax, lc1[v]);
      cmax = fmaxf(cmax, __shfl_xor(cmax, 16, 32));
      const float mnew = fmaxf(m, cmax);
      const float sc   = __expf(m - mnew);           // 0 on first block
      float csum = 0.0f;
      float p0[8], p1[8];
#pragma unroll
      for (int v = 0; v < 8; ++v) {
        p0[v] = __expf(lc0[v] - mnew); csum += p0[v];
        p1[v] = __expf(lc1[v] - mnew); csum += p1[v];
      }
      csum += __shfl_xor(csum, 16, 32);
      lsum = lsum * sc + csum;
      m = mnew;
      // Lane holds s = v + 8*half (tile0) and 16 + v + 8*half (tile1):
      // contiguous 8-half runs in its own t-row of Pl.
#pragma unroll
      for (int v = 0; v < 8; ++v) {
        Pl[l16 * 40 + 8 * half + v]      = (_Float16)p0[v];
        Pl[l16 * 40 + 16 + 8 * half + v] = (_Float16)p1[v];
      }
      if (half == 0) scaleL[l16] = sc;
    }
    __syncthreads();

    const float sc = scaleL[l16];
#pragma unroll
    for (int ot = 0; ot < 4; ++ot)
#pragma unroll
      for (int v = 0; v < 8; ++v) acc[ot][v] *= sc;

    // acc += Q(16o x 32s) @ P(32s x 16t): one K=32 WMMA per o-tile.
    v16h bp = *(const v16h*)(Pl + l16 * 40 + 16 * half);
#pragma unroll
    for (int ot = 0; ot < 4; ++ot) {
      const int orow = wave * 64 + ot * 16 + l16;
      v16h a = load_frag16(Qn + (size_t)orow * S + s0 + 16 * half);
      acc[ot] = __builtin_amdgcn_wmma_f32_16x16x32_f16(false, a, false, bp,
                                                       (short)0, acc[ot],
                                                       false, false);
    }

    if (sb + 1 < nsb) {
      __builtin_prefetch(Kn + (size_t)(s0 + 32 + l16) * R, 0, 1);
      __builtin_prefetch(Qn + (size_t)(wave * 64 + l16) * S + s0 + 32, 0, 1);
    }
    __syncthreads();               // P consumed before wave 0 rewrites it
  }

  if (wave_u == 0) { if (half == 0) linvL[l16] = 1.0f / lsum; }
  __syncthreads();

  const float g    = gamma[0];
  const float linv = linvL[l16];
#pragma unroll
  for (int ot = 0; ot < 4; ++ot) {
#pragma unroll
    for (int v = 0; v < 8; ++v) {
      const int row = wave * 64 + ot * 16 + v + 8 * half;
      const size_t idx = (size_t)row * S + (t0 + l16);
      On[idx] = g * acc[ot][v] * linv + Xn[idx];
    }
  }
}

// ---------------------------------------------------------------------------
extern "C" void kernel_launch(void* const* d_in, const int* in_sizes, int n_in,
                              void* d_out, int out_size, void* d_ws, size_t ws_size,
                              hipStream_t stream) {
  (void)n_in; (void)out_size; (void)ws_size;
  const float* x     = (const float*)d_in[0];
  const float* k_w   = (const float*)d_in[1];
  const float* k_b   = (const float*)d_in[2];
  const float* v_w   = (const float*)d_in[3];
  const float* v_b   = (const float*)d_in[4];
  const float* q_w   = (const float*)d_in[5];
  const float* q_b   = (const float*)d_in[6];
  const float* gamma = (const float*)d_in[7];

  const int R = in_sizes[2];                 // 64
  const int C = in_sizes[6];                 // 512
  const int S = 8 * 16 * 32;                 // 4096
  const int N = in_sizes[0] / (C * S);       // 4

  _Float16* Kws  = (_Float16*)d_ws;                // [N][S][R]
  _Float16* Vws  = Kws  + (size_t)N * S * R;       // [N][S][R] (t-major)
  _Float16* Qws  = Vws  + (size_t)N * S * R;       // [N][C][S]
  _Float16* Xt16 = Qws  + (size_t)N * C * S;       // [N][S][C]
  _Float16* kw16 = Xt16 + (size_t)N * S * C;       // [R][C]
  _Float16* vw16 = kw16 + (size_t)R * C;           // [R][C]
  _Float16* qw16 = vw16 + (size_t)R * C;           // [C][C]
  float* out = (float*)d_out;

  // 1) weights -> f16
  cvt_f16_kernel<<<dim3((R * C + 255) / 256), dim3(256), 0, stream>>>(k_w, kw16, R * C);
  cvt_f16_kernel<<<dim3((R * C + 255) / 256), dim3(256), 0, stream>>>(v_w, vw16, R * C);
  cvt_f16_kernel<<<dim3((C * C + 255) / 256), dim3(256), 0, stream>>>(q_w, qw16, C * C);
  // 2) x -> f16 transposed [S][C]
  transpose_cvt_kernel<<<dim3(S / 32, C / 32, N), dim3(32, 8), 0, stream>>>(
      x, Xt16, C, S);
  // 3) projections (all on the f16 K=32 matrix path)
  proj_gemm16_kernel<<<dim3(S / 128, R / 16, N), dim3(256), 0, stream>>>(
      kw16, k_b, Xt16, Kws, R, C, S, /*transposeOut=*/1);
  proj_gemm16_kernel<<<dim3(S / 128, R / 16, N), dim3(256), 0, stream>>>(
      vw16, v_b, Xt16, Vws, R, C, S, /*transposeOut=*/1);
  proj_gemm16_kernel<<<dim3(S / 128, C / 16, N), dim3(256), 0, stream>>>(
      qw16, q_b, Xt16, Qws, C, C, S, /*transposeOut=*/0);
  // 4) flash attention + residual
  gsa_attn_kernel<<<dim3(S / 16, N), dim3(256), 0, stream>>>(
      Kws, Vws, Qws, x, gamma, out, C, R, S);
}
